// BlockViTStage2_2714419331290
// MI455X (gfx1250) — matrix operand order repaired and checked by reading, not compiled, not run on hardware
//
#include <hip/hip_runtime.h>
#include <hip/hip_bf16.h>
#include <math.h>

// Problem constants (match reference)
#define BB   8
#define NN   1024
#define C2D  512
#define KVD  1792
#define NH   4
#define FFD  2048   // C2*EXPAND
#define MTOT 8192   // B*N

typedef __bf16 bf16_t;
typedef bf16_t v16bf __attribute__((ext_vector_type(16)));
typedef float  v8f   __attribute__((ext_vector_type(8)));
typedef unsigned int v4u __attribute__((ext_vector_type(4)));

union FragAB { v16bf v; v4u q[2]; };

#if defined(__has_builtin)
#if __has_builtin(__builtin_amdgcn_s_wait_asynccnt)
#define WAIT_ASYNC(n) __builtin_amdgcn_s_wait_asynccnt(n)
#endif
#endif
#ifndef WAIT_ASYNC
#define WAIT_ASYNC(n) asm volatile("s_wait_asynccnt %0" :: "i"(n) : "memory")
#endif

__device__ __forceinline__ unsigned short f2bf(float f) {
  unsigned int u = __float_as_uint(f);
  u += 0x7FFFu + ((u >> 16) & 1u);           // round-to-nearest-even
  return (unsigned short)(u >> 16);
}

__device__ __forceinline__ float block_sum(float v, float* sh) {
  int tid = threadIdx.x;
  sh[tid] = v; __syncthreads();
  for (int s = 128; s > 0; s >>= 1) { if (tid < s) sh[tid] += sh[tid + s]; __syncthreads(); }
  float r = sh[0]; __syncthreads();
  return r;
}

__device__ __forceinline__ float block_max(float v, float* sh) {
  int tid = threadIdx.x;
  sh[tid] = v; __syncthreads();
  for (int s = 128; s > 0; s >>= 1) { if (tid < s) sh[tid] = fmaxf(sh[tid], sh[tid + s]); __syncthreads(); }
  float r = sh[0]; __syncthreads();
  return r;
}

// ---------------------------------------------------------------------------
// Block-cooperative WMMA core.
// Block = 256 threads = 8 waves, computes a 128(M) x 64(N) tile.
// Wave w owns rows [m0wave, m0wave+16); all waves share columns [n0, n0+64).
// Per K-step (32): the 64x32 bf16 B tile (4KB) is DMA'd into LDS with
// global_load_async_to_lds_b128 (each thread issues one 16B chunk, i.e. one
// async instruction per wave), double buffered against the WMMAs.
// A fragments stream straight from global (b128 pairs, pipelined by hw cnts).
// Fragment k order: contiguous, k = 16*(lane/16) + s, same for A and B (wave32).
// ---------------------------------------------------------------------------
__device__ __forceinline__ void gemm_block(const unsigned short* __restrict__ A,
                                           int lda, int m0wave,
                                           const unsigned short* __restrict__ Bt,
                                           int ldb, int n0, int K,
                                           v8f acc[4]) {
  __shared__ v4u btile[2][256];              // 2 x (64 rows x 64B) = 8KB
  const int tid  = threadIdx.x;
  const int lane = tid & 31, hlf = lane >> 4, r = lane & 15;
  const int brow = tid >> 2, bchunk = tid & 3;

  const unsigned short* bsrc = Bt + (size_t)(n0 + brow) * ldb + (bchunk << 3);
  const v4u* ap = reinterpret_cast<const v4u*>(A + ((size_t)m0wave + r) * lda);

  // preload chunk 0 into buffer 0
  {
    unsigned int loff = (unsigned int)(unsigned long long)&btile[0][brow * 4 + bchunk];
    asm volatile("global_load_async_to_lds_b128 %0, %1, off"
                 :: "v"(loff), "v"(bsrc) : "memory");
  }

  const int nch = K >> 5;
  for (int c = 0; c < nch; ++c) {
    if (c + 1 < nch) {
      unsigned int loff = (unsigned int)(unsigned long long)&btile[(c + 1) & 1][brow * 4 + bchunk];
      const unsigned short* src = bsrc + ((size_t)(c + 1) << 5);
      asm volatile("global_load_async_to_lds_b128 %0, %1, off"
                   :: "v"(loff), "v"(src) : "memory");
      WAIT_ASYNC(1);                         // current chunk's DMA done (in-order)
    } else {
      WAIT_ASYNC(0);
    }
    __syncthreads();                         // whole 64x32 B tile visible to block

    __builtin_prefetch((const void*)(ap + 8), 0, 3);
    FragAB a;  a.q[0] = ap[2 * hlf]; a.q[1] = ap[2 * hlf + 1];
    const v4u* bt = &btile[c & 1][0];
    FragAB b0; b0.q[0] = bt[(r     ) * 4 + 2 * hlf]; b0.q[1] = bt[(r     ) * 4 + 2 * hlf + 1];
    FragAB b1; b1.q[0] = bt[(r + 16) * 4 + 2 * hlf]; b1.q[1] = bt[(r + 16) * 4 + 2 * hlf + 1];
    FragAB b2; b2.q[0] = bt[(r + 32) * 4 + 2 * hlf]; b2.q[1] = bt[(r + 32) * 4 + 2 * hlf + 1];
    FragAB b3; b3.q[0] = bt[(r + 48) * 4 + 2 * hlf]; b3.q[1] = bt[(r + 48) * 4 + 2 * hlf + 1];
    acc[0] = __builtin_amdgcn_wmma_f32_16x16x32_bf16(false, a.v, false, b0.v, (short)0, acc[0], false, false);
    acc[1] = __builtin_amdgcn_wmma_f32_16x16x32_bf16(false, a.v, false, b1.v, (short)0, acc[1], false, false);
    acc[2] = __builtin_amdgcn_wmma_f32_16x16x32_bf16(false, a.v, false, b2.v, (short)0, acc[2], false, false);
    acc[3] = __builtin_amdgcn_wmma_f32_16x16x32_bf16(false, a.v, false, b3.v, (short)0, acc[3], false, false);
    ap += 4;
    __syncthreads();                         // retire buffer before next DMA overwrites it
  }
}

__device__ __forceinline__ void zero_acc(v8f acc[4]) {
#pragma unroll
  for (int t = 0; t < 4; ++t) acc[t] = v8f{0.f,0.f,0.f,0.f,0.f,0.f,0.f,0.f};
}

// ---------------------------------------------------------------------------
// Elementwise / reduction kernels (blockDim == 256)
// ---------------------------------------------------------------------------
__global__ void concat_ln_kernel(const float* __restrict__ e1, const float* __restrict__ e2,
                                 const float* __restrict__ e3, const float* __restrict__ g,
                                 const float* __restrict__ bta, unsigned short* __restrict__ out) {
  __shared__ float sh[256];
  const int t = blockIdx.x, tid = threadIdx.x;
  float x[7]; float s = 0.f;
#pragma unroll
  for (int j = 0; j < 7; ++j) {
    int i = tid + j * 256;
    float v;
    if (i < 256)      v = e1[(size_t)t * 256  + i];
    else if (i < 768) v = e2[(size_t)t * 512  + (i - 256)];
    else              v = e3[(size_t)t * 1024 + (i - 768)];
    x[j] = v; s += v;
  }
  float mean = block_sum(s, sh) * (1.0f / 1792.0f);
  float ss = 0.f;
#pragma unroll
  for (int j = 0; j < 7; ++j) { float d = x[j] - mean; ss += d * d; }
  float var = block_sum(ss, sh) * (1.0f / 1792.0f);
  float rstd = rsqrtf(var + 1e-6f);
#pragma unroll
  for (int j = 0; j < 7; ++j) {
    int i = tid + j * 256;
    out[(size_t)t * 1792 + i] = f2bf((x[j] - mean) * rstd * g[i] + bta[i]);
  }
}

__global__ void row_ln_kernel(const float* __restrict__ x, const float* __restrict__ g,
                              const float* __restrict__ bta, unsigned short* __restrict__ y, int len) {
  __shared__ float sh[256];
  const int row = blockIdx.x, tid = threadIdx.x;
  const float* p = x + (size_t)row * len;
  float s = 0.f;
  for (int i = tid; i < len; i += 256) s += p[i];
  float mean = block_sum(s, sh) / (float)len;
  float ss = 0.f;
  for (int i = tid; i < len; i += 256) { float d = p[i] - mean; ss += d * d; }
  float var = block_sum(ss, sh) / (float)len;
  float rstd = rsqrtf(var + 1e-6f);
  for (int i = tid; i < len; i += 256)
    y[(size_t)row * len + i] = f2bf((p[i] - mean) * rstd * g[i] + bta[i]);
}

// wt[batch][o][i] = bf16(w[batch][i][o])
__global__ void wtrans_kernel(const float* __restrict__ w, unsigned short* __restrict__ wt,
                              int I, int O) {
  size_t base = (size_t)blockIdx.y * I * O;
  size_t total = (size_t)I * O;
  for (size_t idx = (size_t)blockIdx.x * 256 + threadIdx.x; idx < total; idx += (size_t)gridDim.x * 256) {
    size_t i = idx % (size_t)I, o = idx / (size_t)I;
    wt[base + idx] = f2bf(w[base + i * O + o]);
  }
}

// instance-norm stats over each [C2 x KV] map
__global__ void inorm_stats_kernel(const float* __restrict__ S, float* __restrict__ stats) {
  __shared__ float sh[256];
  const size_t M = (size_t)C2D * KVD;
  const float* p = S + (size_t)blockIdx.x * M;
  float s = 0.f, ss = 0.f;
  for (size_t i = threadIdx.x; i < M; i += 256) { float v = p[i]; s += v; ss += v * v; }
  float mean = block_sum(s, sh) / (float)M;
  float var  = block_sum(ss, sh) / (float)M - mean * mean;
  if (threadIdx.x == 0) {
    stats[2 * blockIdx.x]     = mean;
    stats[2 * blockIdx.x + 1] = rsqrtf(fmaxf(var, 0.f) + 1e-5f);
  }
}

// per-row: normalize with instance stats, then softmax over 1792, emit bf16
__global__ void softmax_kernel(const float* __restrict__ S, const float* __restrict__ stats,
                               unsigned short* __restrict__ probs) {
  __shared__ float sh[256];
  const int row = blockIdx.x, tid = threadIdx.x;
  const int bh = row >> 9;                     // row / 512
  const float mean = stats[2 * bh], rstd = stats[2 * bh + 1];
  const float* p = S + (size_t)row * KVD;
  float x[7]; float m = -3.4e38f;
#pragma unroll
  for (int j = 0; j < 7; ++j) { x[j] = (p[tid + j * 256] - mean) * rstd; m = fmaxf(m, x[j]); }
  m = block_max(m, sh);
  float s = 0.f;
#pragma unroll
  for (int j = 0; j < 7; ++j) { x[j] = __expf(x[j] - m); s += x[j]; }
  s = block_sum(s, sh);
  float inv = 1.0f / s;
#pragma unroll
  for (int j = 0; j < 7; ++j) probs[(size_t)row * KVD + tid + j * 256] = f2bf(x[j] * inv);
}

// ---------------------------------------------------------------------------
// GEMM kernels: block = 8 waves = 128x64 tile; grid.x = (M/128)*(N/64)
// ---------------------------------------------------------------------------
__global__ void gemm_proj_kernel(const unsigned short* __restrict__ X,
                                 const unsigned short* __restrict__ WT,
                                 unsigned short* __restrict__ Y,
                                 int Cin, int Cout, int trans) {
  const int b = blockIdx.z, h = blockIdx.y;
  const int wave = threadIdx.x >> 5, lane = threadIdx.x & 31;
  const int MB = NN / 128;
  const int mb = blockIdx.x % MB, nb = blockIdx.x / MB;
  const int m0 = mb * 128 + wave * 16, n0 = nb * 64;
  const unsigned short* A  = X  + (size_t)b * NN * Cin;
  const unsigned short* Bt = WT + (size_t)h * Cin * Cout;
  v8f acc[4]; zero_acc(acc);
  gemm_block(A, Cin, m0, Bt, Cin, n0, Cin, acc);
  const int hlf = lane >> 4, r = lane & 15;
  const size_t bh = (size_t)(b * NH + h);
  if (trans) {
#pragma unroll
    for (int t = 0; t < 4; ++t)
#pragma unroll
      for (int j = 0; j < 8; ++j) {
        int col = n0 + 16 * t + r, row = m0 + j + 8 * hlf;
        Y[(bh * Cout + col) * NN + row] = f2bf(acc[t][j]);
      }
  } else {
#pragma unroll
    for (int t = 0; t < 4; ++t)
#pragma unroll
      for (int j = 0; j < 8; ++j) {
        int col = n0 + 16 * t + r, row = m0 + j + 8 * hlf;
        Y[(bh * NN + row) * Cout + col] = f2bf(acc[t][j]);
      }
  }
}

__global__ void gemm_scores_kernel(const unsigned short* __restrict__ qt,
                                   const unsigned short* __restrict__ kt,
                                   float* __restrict__ S, float scale) {
  const int bh = blockIdx.y;
  const int wave = threadIdx.x >> 5, lane = threadIdx.x & 31;
  const int MB = C2D / 128;
  const int mb = blockIdx.x % MB, nb = blockIdx.x / MB;
  const int m0 = mb * 128 + wave * 16, n0 = nb * 64;
  const unsigned short* A  = qt + (size_t)bh * C2D * NN;
  const unsigned short* Bt = kt + (size_t)bh * KVD * NN;
  v8f acc[4]; zero_acc(acc);
  gemm_block(A, NN, m0, Bt, NN, n0, NN, acc);
  const int hlf = lane >> 4, r = lane & 15;
#pragma unroll
  for (int t = 0; t < 4; ++t)
#pragma unroll
    for (int j = 0; j < 8; ++j) {
      int col = n0 + 16 * t + r, row = m0 + j + 8 * hlf;
      S[((size_t)bh * C2D + row) * KVD + col] = acc[t][j] * scale;
    }
}

__global__ void gemm_ctx_kernel(const unsigned short* __restrict__ probs,
                                const unsigned short* __restrict__ vv,
                                unsigned short* __restrict__ ctx) {
  const int b = blockIdx.y;
  const int wave = threadIdx.x >> 5, lane = threadIdx.x & 31;
  const int MB = C2D / 128;
  const int mb = blockIdx.x % MB, nb = blockIdx.x / MB;
  const int m0 = mb * 128 + wave * 16, n0 = nb * 64;
  v8f acc[4]; zero_acc(acc);
  for (int h = 0; h < NH; ++h) {
    const unsigned short* A  = probs + ((size_t)(b * NH + h)) * C2D * KVD;
    const unsigned short* Bt = vv    + ((size_t)(b * NH + h)) * NN * KVD;
    gemm_block(A, KVD, m0, Bt, KVD, n0, KVD, acc);
  }
  const int hlf = lane >> 4, r = lane & 15;
#pragma unroll
  for (int t = 0; t < 4; ++t)
#pragma unroll
    for (int j = 0; j < 8; ++j) {
      int tok = n0 + 16 * t + r, c = m0 + j + 8 * hlf;
      ctx[((size_t)b * NN + tok) * C2D + c] = f2bf(acc[t][j] * 0.25f);  // mean over heads
    }
}

__global__ void gemm_o_kernel(const unsigned short* __restrict__ ctx,
                              const unsigned short* __restrict__ WoutT,
                              const float* __restrict__ emb2, float* __restrict__ cx2) {
  const int wave = threadIdx.x >> 5, lane = threadIdx.x & 31;
  const int MB = MTOT / 128;
  const int mb = blockIdx.x % MB, nb = blockIdx.x / MB;
  const int m0 = mb * 128 + wave * 16, n0 = nb * 64;
  v8f acc[4]; zero_acc(acc);
  gemm_block(ctx, C2D, m0, WoutT, C2D, n0, C2D, acc);
  const int hlf = lane >> 4, r = lane & 15;
#pragma unroll
  for (int t = 0; t < 4; ++t)
#pragma unroll
    for (int j = 0; j < 8; ++j) {
      int col = n0 + 16 * t + r, row = m0 + j + 8 * hlf;
      size_t o = (size_t)row * C2D + col;
      cx2[o] = emb2[o] + acc[t][j];
    }
}

__global__ void gemm_fc1_kernel(const unsigned short* __restrict__ x2n,
                                const unsigned short* __restrict__ fc1T,
                                const float* __restrict__ bias, unsigned short* __restrict__ h1) {
  const int wave = threadIdx.x >> 5, lane = threadIdx.x & 31;
  const int MB = MTOT / 128;
  const int mb = blockIdx.x % MB, nb = blockIdx.x / MB;
  const int m0 = mb * 128 + wave * 16, n0 = nb * 64;
  v8f acc[4]; zero_acc(acc);
  gemm_block(x2n, C2D, m0, fc1T, C2D, n0, C2D, acc);
  const int hlf = lane >> 4, r = lane & 15;
#pragma unroll
  for (int t = 0; t < 4; ++t)
#pragma unroll
    for (int j = 0; j < 8; ++j) {
      int col = n0 + 16 * t + r, row = m0 + j + 8 * hlf;
      float v = acc[t][j] + bias[col];
      float g = 0.5f * v * (1.0f + erff(v * 0.70710678118654752f));   // exact GELU
      h1[(size_t)row * FFD + col] = f2bf(g);
    }
}

__global__ void gemm_fc2_kernel(const unsigned short* __restrict__ h1,
                                const unsigned short* __restrict__ fc2T,
                                const float* __restrict__ bias, const float* __restrict__ cx2,
                                float* __restrict__ out) {
  const int wave = threadIdx.x >> 5, lane = threadIdx.x & 31;
  const int MB = MTOT / 128;
  const int mb = blockIdx.x % MB, nb = blockIdx.x / MB;
  const int m0 = mb * 128 + wave * 16, n0 = nb * 64;
  v8f acc[4]; zero_acc(acc);
  gemm_block(h1, FFD, m0, fc2T, FFD, n0, FFD, acc);
  const int hlf = lane >> 4, r = lane & 15;
#pragma unroll
  for (int t = 0; t < 4; ++t)
#pragma unroll
    for (int j = 0; j < 8; ++j) {
      int col = n0 + 16 * t + r, row = m0 + j + 8 * hlf;
      size_t o = (size_t)row * C2D + col;
      out[o] = acc[t][j] + bias[col] + cx2[o];
    }
}

// ---------------------------------------------------------------------------
extern "C" void kernel_launch(void* const* d_in, const int* in_sizes, int n_in,
                              void* d_out, int out_size, void* d_ws, size_t ws_size,
                              hipStream_t stream) {
  (void)in_sizes; (void)n_in; (void)out_size; (void)ws_size;
  const float* emb1   = (const float*)d_in[0];
  const float* emb2   = (const float*)d_in[1];
  const float* emb3   = (const float*)d_in[2];
  const float* Wq     = (const float*)d_in[3];
  const float* Wk     = (const float*)d_in[4];
  const float* Wv     = (const float*)d_in[5];
  const float* Wout   = (const float*)d_in[6];
  const float* ln1_g  = (const float*)d_in[7];
  const float* ln1_b  = (const float*)d_in[8];
  const float* lnall_g= (const float*)d_in[9];
  const float* lnall_b= (const float*)d_in[10];
  const float* lnffn_g= (const float*)d_in[11];
  const float* lnffn_b= (const float*)d_in[12];
  const float* fc1_w  = (const float*)d_in[13];
  const float* fc1_b  = (const float*)d_in[14];
  const float* fc2_w  = (const float*)d_in[15];
  const float* fc2_b  = (const float*)d_in[16];

  char* p = (char*)d_ws;
  auto alloc = [&](size_t bytes) -> void* {
    void* r = (void*)p;
    p += (bytes + 255) & ~(size_t)255;
    return r;
  };

  unsigned short* cx2n  = (unsigned short*)alloc((size_t)MTOT * C2D * 2);
  unsigned short* kvn   = (unsigned short*)alloc((size_t)MTOT * KVD * 2);
  unsigned short* WqT   = (unsigned short*)alloc((size_t)NH * C2D * C2D * 2);
  unsigned short* WkT   = (unsigned short*)alloc((size_t)NH * KVD * KVD * 2);
  unsigned short* WvT   = (unsigned short*)alloc((size_t)NH * KVD * KVD * 2);
  unsigned short* WoutT = (unsigned short*)alloc((size_t)C2D * C2D * 2);
  unsigned short* fc1T  = (unsigned short*)alloc((size_t)C2D * FFD * 2);
  unsigned short* fc2T  = (unsigned short*)alloc((size_t)C2D * FFD * 2);
  unsigned short* qt    = (unsigned short*)alloc((size_t)BB * NH * C2D * NN * 2);
  unsigned short* kt    = (unsigned short*)alloc((size_t)BB * NH * KVD * NN * 2);
  unsigned short* vv    = (unsigned short*)alloc((size_t)BB * NH * NN * KVD * 2);
  float*          S     = (float*)alloc((size_t)BB * NH * C2D * KVD * 4);
  float*          stats = (float*)alloc((size_t)BB * NH * 2 * 4);
  unsigned short* probs = (unsigned short*)alloc((size_t)BB * NH * C2D * KVD * 2);
  unsigned short* ctx   = (unsigned short*)alloc((size_t)MTOT * C2D * 2);
  float*          cx2   = (float*)alloc((size_t)MTOT * C2D * 4);
  unsigned short* x2n   = (unsigned short*)alloc((size_t)MTOT * C2D * 2);
  unsigned short* h1    = (unsigned short*)alloc((size_t)MTOT * FFD * 2);

  // 1) LayerNorms on activations -> bf16
  concat_ln_kernel<<<MTOT, 256, 0, stream>>>(emb1, emb2, emb3, lnall_g, lnall_b, kvn);
  row_ln_kernel<<<MTOT, 256, 0, stream>>>(emb2, ln1_g, ln1_b, cx2n, C2D);

  // 2) Weights -> bf16 transposed [out,in]
  wtrans_kernel<<<dim3((C2D * C2D + 255) / 256, NH), 256, 0, stream>>>(Wq, WqT, C2D, C2D);
  wtrans_kernel<<<dim3((KVD * KVD + 255) / 256, NH), 256, 0, stream>>>(Wk, WkT, KVD, KVD);
  wtrans_kernel<<<dim3((KVD * KVD + 255) / 256, NH), 256, 0, stream>>>(Wv, WvT, KVD, KVD);
  wtrans_kernel<<<dim3((C2D * C2D + 255) / 256, 1), 256, 0, stream>>>(Wout, WoutT, C2D, C2D);
  wtrans_kernel<<<dim3((C2D * FFD + 255) / 256, 1), 256, 0, stream>>>(fc1_w, fc1T, C2D, FFD);
  wtrans_kernel<<<dim3((FFD * C2D + 255) / 256, 1), 256, 0, stream>>>(fc2_w, fc2T, FFD, C2D);

  // 3) Per-head projections (q/k transposed outputs, v natural)
  gemm_proj_kernel<<<dim3((NN/128)*(C2D/64), NH, BB), 256, 0, stream>>>(cx2n, WqT, qt, C2D, C2D, 1);
  gemm_proj_kernel<<<dim3((NN/128)*(KVD/64), NH, BB), 256, 0, stream>>>(kvn, WkT, kt, KVD, KVD, 1);
  gemm_proj_kernel<<<dim3((NN/128)*(KVD/64), NH, BB), 256, 0, stream>>>(kvn, WvT, vv, KVD, KVD, 0);

  // 4) Channel-attention scores, instance-norm, softmax
  const float scale = 1.0f / sqrtf((float)KVD);
  gemm_scores_kernel<<<dim3((C2D/128)*(KVD/64), BB * NH), 256, 0, stream>>>(qt, kt, S, scale);
  inorm_stats_kernel<<<BB * NH, 256, 0, stream>>>(S, stats);
  softmax_kernel<<<BB * NH * C2D, 256, 0, stream>>>(S, stats, probs);

  // 5) Context (mean over heads), out projection + residual
  gemm_ctx_kernel<<<dim3((C2D/128)*(NN/64), BB), 256, 0, stream>>>(probs, vv, ctx);
  gemm_o_kernel<<<(MTOT/128)*(C2D/64), 256, 0, stream>>>(ctx, WoutT, emb2, cx2);

  // 6) FFN with residual
  row_ln_kernel<<<MTOT, 256, 0, stream>>>(cx2, lnffn_g, lnffn_b, x2n, C2D);
  gemm_fc1_kernel<<<(MTOT/128)*(FFD/64), 256, 0, stream>>>(x2n, fc1T, fc1_b, h1);
  gemm_fc2_kernel<<<(MTOT/128)*(C2D/64), 256, 0, stream>>>(h1, fc2T, fc2_b, cx2, (float*)d_out);
}
